// SelfAttention_24739011625513
// MI455X (gfx1250) — matrix operand order repaired and checked by reading, not compiled, not run on hardware
//
#include <hip/hip_runtime.h>

typedef _Float16 f16;
typedef _Float16 f16x2 __attribute__((ext_vector_type(2)));
typedef _Float16 f16x8 __attribute__((ext_vector_type(8)));
typedef _Float16 v16h  __attribute__((ext_vector_type(16)));
typedef float    v8f   __attribute__((ext_vector_type(8)));

#define B_  8
#define C_  256
#define N_  4096
#define KC_ 32
#define VC_ 256
#define LOG2E_ 1.44269504088896340736f

#if __has_builtin(__builtin_amdgcn_exp2f)
__device__ __forceinline__ float fast_exp2(float x) { return __builtin_amdgcn_exp2f(x); }
#else
__device__ __forceinline__ float fast_exp2(float x) { return exp2f(x); }
#endif

// v_permlane16_b32 butterfly helper: pure-VALU xor-shuffle within 16 lanes
__device__ __forceinline__ float permlane16_f(float x, int sl, int sh) {
  return __int_as_float(__builtin_amdgcn_permlane16(
      __float_as_int(x), __float_as_int(x), sl, sh, false, false));
}
__device__ __forceinline__ float rowmax16(float x) {
  x = fmaxf(x, permlane16_f(x, (int)0x67452301, (int)0xEFCDAB89)); // xor 1
  x = fmaxf(x, permlane16_f(x, (int)0x54761032, (int)0xDCFE98BA)); // xor 2
  x = fmaxf(x, permlane16_f(x, (int)0x32107654, (int)0xBA98FEDC)); // xor 4
  x = fmaxf(x, permlane16_f(x, (int)0xFEDCBA98, (int)0x76543210)); // xor 8
  return x;
}

// -------------------------------------------------------------------------
// Stage 1: fused 1x1-conv projections  x[B,C,N] -> f16 Q[b][n][32] (pre-
// scaled by log2(e) so the attention softmax runs in exp2 domain),
// Kt[b][n][32], V[b][vc][n_perm] (per-32-key interleave perm(16t+c)=2c+t
// matching the attention kernel's packed P stores).
// Each thread keeps 8 channel accumulators so one LDS read of x feeds 8
// FMAs (8x less DS traffic than one-read-per-FMA).
// -------------------------------------------------------------------------
__global__ void __launch_bounds__(256) proj_kernel(
    const float* __restrict__ x,
    const float* __restrict__ Wq, const float* __restrict__ bq,
    const float* __restrict__ Wk, const float* __restrict__ bk,
    const float* __restrict__ Wv, const float* __restrict__ bv,
    f16* __restrict__ Qh, f16* __restrict__ Kth, f16* __restrict__ Vh)
{
  __shared__ float xs[C_ * 32];                // 32 KB: x[b, :, n0:n0+32]
  const int blk = blockIdx.x;                  // B * (N/32) blocks
  const int b   = blk >> 7;                    // 128 blocks per batch
  const int n0  = (blk & 127) * 32;
  const int tid = threadIdx.x;

  const float* xb = x + ((size_t)b * C_) * N_ + n0;
  #pragma unroll
  for (int k = 0; k < (C_ * 32) / 256; ++k) {
    int idx = tid + k * 256;
    int c = idx >> 5, i = idx & 31;
    xs[idx] = xb[(size_t)c * N_ + i];
  }
  __syncthreads();

  const int p    = tid & 31;                   // pixel within chunk
  const int cgrp = tid >> 5;                   // 0..7 (wave-uniform)
  const int n    = n0 + p;
  const int pperm = ((p & 15) << 1) | (p >> 4); // key interleave for V

  for (int g = 0; g < 5; ++g) {                // 5 groups x 8 channels
    const float* wp[8]; float acc8[8];
    #pragma unroll
    for (int i = 0; i < 8; ++i) {
      int ch = cgrp + (g << 6) + (i << 3);     // 0..319, wave-uniform
      if (ch < KC_)          { wp[i] = Wq + ch * C_;             acc8[i] = bq[ch]; }
      else if (ch < 2 * KC_) { wp[i] = Wk + (ch - KC_) * C_;     acc8[i] = bk[ch - KC_]; }
      else                   { wp[i] = Wv + (ch - 2 * KC_) * C_; acc8[i] = bv[ch - 2 * KC_]; }
    }
    #pragma unroll 4
    for (int c = 0; c < C_; ++c) {
      float xv = xs[(c << 5) + p];             // 1 DS read -> 8 FMAs
      #pragma unroll
      for (int i = 0; i < 8; ++i) acc8[i] += wp[i][c] * xv;
    }
    #pragma unroll
    for (int i = 0; i < 8; ++i) {
      int ch = cgrp + (g << 6) + (i << 3);
      if (ch < KC_)          Qh [((size_t)b * N_ + n) * KC_ + ch] = (f16)(acc8[i] * LOG2E_);
      else if (ch < 2 * KC_) Kth[((size_t)b * N_ + n) * KC_ + (ch - KC_)] = (f16)acc8[i];
      else                   Vh [((size_t)b * VC_ + (ch - 2 * KC_)) * N_ + n0 + pperm] = (f16)acc8[i];
    }
  }
}

// -------------------------------------------------------------------------
// Stage 2: flash-attention, 64 keys per iteration.
//   4 WMMA  : S = Q.K^T (Q pre-scaled by log2e -> softmax via exp2)
//   VALU    : row-max via v_permlane16 butterflies (no LDS traffic)
//   2 WMMA  : row-sum(P) = P x ones  (replicated into C-layout rows)
//   16 WMMA : O += P.V  (8 column tiles x 2 key halves)
// P transpose C->A layout via per-wave LDS with packed b32 stores.
// -------------------------------------------------------------------------
__global__ void __launch_bounds__(256) attn_kernel(
    const f16* __restrict__ Qh, const f16* __restrict__ Kth,
    const f16* __restrict__ Vh, const float* __restrict__ x,
    const float* __restrict__ gamma, float* __restrict__ out)
{
  __shared__ f16 Pbuf[8][16][64];              // 16 KB, private per wave
  const int b    = blockIdx.x >> 6;
  const int m0   = (blockIdx.x & 63) << 6;
  const int tid  = threadIdx.x;
  const int wave = tid >> 5, lane = tid & 31;
  const int r = wave >> 1, h = wave & 1;
  const int half = lane >> 4, lm = lane & 15;
  const int m_tile  = m0 + (r << 4);
  const int colbase = h << 7;

  // Q A-fragment (16x32 f16): lane lm = row, K = 16*half + s (contiguous)
  const f16* qrow = Qh + ((size_t)b * N_ + (m_tile + lm)) * KC_ + 16 * half;
  f16x8 qlo = *(const f16x8*)qrow;
  f16x8 qhi = *(const f16x8*)(qrow + 8);
  v16h Qa = __builtin_shufflevector(qlo, qhi,
      0,1,2,3,4,5,6,7,8,9,10,11,12,13,14,15);

  v16h onesB;
  #pragma unroll
  for (int i = 0; i < 16; ++i) onesB[i] = (f16)1.0f;

  v8f zero = {};
  v8f acc[8];
  float m_run[8], l_run[8];
  #pragma unroll
  for (int t = 0; t < 8; ++t) acc[t] = zero;
  #pragma unroll
  for (int v = 0; v < 8; ++v) { m_run[v] = -1e30f; l_run[v] = 0.0f; }

  for (int j = 0; j < N_; j += 64) {
    // ---- S = Q.K^T over 64 keys (4 tiles) ----
    const f16* kp = Kth + ((size_t)b * N_ + (j + lm)) * KC_ + 16 * half;
    __builtin_prefetch(kp + 64 * KC_, 0, 3);   // next K block
    v16h Kb0 = *(const v16h*)kp;
    v16h Kb1 = *(const v16h*)(kp + 16 * KC_);
    v16h Kb2 = *(const v16h*)(kp + 32 * KC_);
    v16h Kb3 = *(const v16h*)(kp + 48 * KC_);
    v8f S0 = __builtin_amdgcn_wmma_f32_16x16x32_f16(false, Qa, false, Kb0, (short)0, zero, false, false);
    v8f S1 = __builtin_amdgcn_wmma_f32_16x16x32_f16(false, Qa, false, Kb1, (short)0, zero, false, false);
    v8f S2 = __builtin_amdgcn_wmma_f32_16x16x32_f16(false, Qa, false, Kb2, (short)0, zero, false, false);
    v8f S3 = __builtin_amdgcn_wmma_f32_16x16x32_f16(false, Qa, false, Kb3, (short)0, zero, false, false);

    // ---- online softmax in exp2 domain ----
    float sc[8];
    #pragma unroll
    for (int v = 0; v < 8; ++v) {
      float mx = fmaxf(fmaxf(S0[v], S1[v]), fmaxf(S2[v], S3[v]));
      mx = rowmax16(mx);
      float mn = fmaxf(m_run[v], mx);
      sc[v] = fast_exp2(m_run[v] - mn);
      m_run[v] = mn;
      float p0 = fast_exp2(S0[v] - mn);
      float p1 = fast_exp2(S1[v] - mn);
      float p2 = fast_exp2(S2[v] - mn);
      float p3 = fast_exp2(S3[v] - mn);
      int row = v + 8 * half;
      f16x2 pk01 = {(f16)p0, (f16)p1};
      f16x2 pk23 = {(f16)p2, (f16)p3};
      *(f16x2*)&Pbuf[wave][row][2 * lm]      = pk01;   // keys j   + {lm, lm+16}
      *(f16x2*)&Pbuf[wave][row][32 + 2 * lm] = pk23;   // keys j+32+ {lm, lm+16}
    }
    asm volatile("" ::: "memory");

    // ---- P: LDS -> A-layout fragments (two 16x32 tiles) ----
    const f16* prow = &Pbuf[wave][lm][0];
    f16x8 a0 = *(const f16x8*)(prow + 16 * half);
    f16x8 a1 = *(const f16x8*)(prow + 8 + 16 * half);
    f16x8 a2 = *(const f16x8*)(prow + 32 + 16 * half);
    f16x8 a3 = *(const f16x8*)(prow + 40 + 16 * half);
    v16h Pa0 = __builtin_shufflevector(a0, a1, 0,1,2,3,4,5,6,7,8,9,10,11,12,13,14,15);
    v16h Pa1 = __builtin_shufflevector(a2, a3, 0,1,2,3,4,5,6,7,8,9,10,11,12,13,14,15);
    asm volatile("" ::: "memory");

    // ---- row-sum(P) by WMMA against ones; update l, rescale O ----
    v8f lb = __builtin_amdgcn_wmma_f32_16x16x32_f16(false, Pa0, false, onesB, (short)0, zero, false, false);
    lb     = __builtin_amdgcn_wmma_f32_16x16x32_f16(false, Pa1, false, onesB, (short)0, lb,   false, false);
    #pragma unroll
    for (int v = 0; v < 8; ++v) {
      l_run[v] = l_run[v] * sc[v] + lb[v];
      #pragma unroll
      for (int t = 0; t < 8; ++t) acc[t][v] *= sc[v];
    }

    // ---- O += P.V : 8 column tiles x 2 key halves ----
    #pragma unroll
    for (int t = 0; t < 8; ++t) {
      int vc = colbase + (t << 4) + lm;
      const f16* vp = Vh + ((size_t)b * VC_ + vc) * N_ + j + 16 * half;
      v16h Vb0 = *(const v16h*)vp;
      v16h Vb1 = *(const v16h*)(vp + 32);
      acc[t] = __builtin_amdgcn_wmma_f32_16x16x32_f16(false, Pa0, false, Vb0, (short)0, acc[t], false, false);
      acc[t] = __builtin_amdgcn_wmma_f32_16x16x32_f16(false, Pa1, false, Vb1, (short)0, acc[t], false, false);
    }
  }

  float inv[8];
  #pragma unroll
  for (int v = 0; v < 8; ++v) inv[v] = 1.0f / l_run[v];
  const float g = gamma[0];
  const int mbase = m_tile + 8 * half;         // VGPR v -> pixel mbase+v

  #pragma unroll
  for (int t = 0; t < 8; ++t) {
    int vc = colbase + (t << 4) + lm;
    size_t idx = ((size_t)b * VC_ + vc) * N_ + mbase;
    float4 x0 = *(const float4*)(x + idx);
    float4 x1 = *(const float4*)(x + idx + 4);
    float4 o0, o1;
    o0.x = g * acc[t][0] * inv[0] + x0.x;
    o0.y = g * acc[t][1] * inv[1] + x0.y;
    o0.z = g * acc[t][2] * inv[2] + x0.z;
    o0.w = g * acc[t][3] * inv[3] + x0.w;
    o1.x = g * acc[t][4] * inv[4] + x1.x;
    o1.y = g * acc[t][5] * inv[5] + x1.y;
    o1.z = g * acc[t][6] * inv[6] + x1.z;
    o1.w = g * acc[t][7] * inv[7] + x1.w;
    *(float4*)(out + idx)     = o0;
    *(float4*)(out + idx + 4) = o1;
  }
}

extern "C" void kernel_launch(void* const* d_in, const int* in_sizes, int n_in,
                              void* d_out, int out_size, void* d_ws, size_t ws_size,
                              hipStream_t stream) {
  (void)in_sizes; (void)n_in; (void)out_size; (void)ws_size;
  const float* x     = (const float*)d_in[0];
  const float* Wq    = (const float*)d_in[1];
  const float* bq    = (const float*)d_in[2];
  const float* Wk    = (const float*)d_in[3];
  const float* bk    = (const float*)d_in[4];
  const float* Wv    = (const float*)d_in[5];
  const float* bv    = (const float*)d_in[6];
  const float* gamma = (const float*)d_in[7];
  float* out = (float*)d_out;

  // workspace: Qh 2MB | Kth 2MB | Vh 16MB  (20 MB total, f16)
  f16* Qh  = (f16*)d_ws;
  f16* Kth = Qh  + (size_t)B_ * N_ * KC_;
  f16* Vh  = Kth + (size_t)B_ * N_ * KC_;

  proj_kernel<<<B_ * (N_ / 32), 256, 0, stream>>>(x, Wq, bq, Wk, bk, Wv, bv,
                                                  Qh, Kth, Vh);
  attn_kernel<<<B_ * (N_ / 64), 256, 0, stream>>>(Qh, Kth, Vh, x, gamma, out);
}
